// SelfAttention_78048145703025
// MI455X (gfx1250) — matrix operand order repaired and checked by reading, not compiled, not run on hardware
//
#include <hip/hip_runtime.h>

typedef __attribute__((ext_vector_type(16))) __bf16 v16bf;
typedef __attribute__((ext_vector_type(8)))  float  v8f;

union Frag { uint4 u[2]; v16bf v; };

#if __has_builtin(__builtin_amdgcn_tanh_f32)
__device__ __forceinline__ float fast_tanh(float x) { return __builtin_amdgcn_tanh_f32(x); }
#elif __has_builtin(__builtin_amdgcn_tanhf)
__device__ __forceinline__ float fast_tanh(float x) { return __builtin_amdgcn_tanhf(x); }
#else
__device__ __forceinline__ float fast_tanh(float x) { return tanhf(x); }
#endif

__device__ __forceinline__ unsigned short f2bf(float f) {
    unsigned u = __float_as_uint(f);
    unsigned r = u + 0x7FFFu + ((u >> 16) & 1u);   // round-to-nearest-even
    return (unsigned short)(r >> 16);
}
__device__ __forceinline__ unsigned pack_bf16(float a, float b) {
    return (unsigned)f2bf(a) | ((unsigned)f2bf(b) << 16);
}

// ---------------------------------------------------------------- W -> bf16
__global__ void __launch_bounds__(256) wconv_kernel(const float* __restrict__ W,
                                                    unsigned short* __restrict__ Wb) {
    const int i = (blockIdx.x * 256 + threadIdx.x) * 4;       // 1M elements total
    float4 v = *(const float4*)(W + i);
    *(uint2*)(Wb + i) = make_uint2(pack_bf16(v.x, v.y), pack_bf16(v.z, v.w));
}

__global__ void __launch_bounds__(256) zero_kernel(float* __restrict__ p, int n) {
    const int i = blockIdx.x * 256 + threadIdx.x;
    if (i < n) p[i] = 0.0f;
}

// ------------------------------------------------- fused GEMM+tanh+key·score
// scores[r] = sum_o key[o] * tanh( sum_h x[r,h]*W[o,h] + bias[o] )
// One workgroup = 32 rows of x. 8 waves: wave>>2 selects the 16-row M-tile,
// wave&3 selects a 256-wide slice of o. LDS x-tile rows padded to 1032 bf16
// (stride 2064 B) so 16 lanes reading 16 rows at equal k are bank-conflict-free.
// K loop is software-pipelined: next step's A/B fragments are in flight while
// the current WMMA executes (breaks the s_wait_loadcnt_dscnt 0 -> wmma chain).
#define XROW 1032

__global__ void __launch_bounds__(256) fused_scores_kernel(
    const float* __restrict__ x,
    const unsigned short* __restrict__ Wb,
    const float* __restrict__ bias,
    const float* __restrict__ keyv,
    float* __restrict__ scores)
{
    extern __shared__ unsigned short xt[];          // 32*1032 bf16 + slack
    const int tid  = threadIdx.x;
    const int lane = tid & 31;
    const int wave = tid >> 5;
    const long rowbase = (long)blockIdx.x * 32;

    // Stage 32 rows of x into LDS as bf16
    {
        const float* xp = x + rowbase * 1024;
        const int col = tid * 4;
        #pragma unroll 4
        for (int r = 0; r < 32; ++r) {
            float4 v = *(const float4*)(xp + (long)r * 1024 + col);
            *(uint2*)&xt[r * XROW + col] =
                make_uint2(pack_bf16(v.x, v.y), pack_bf16(v.z, v.w));
        }
    }
    __syncthreads();

    const int mtile = wave >> 2;            // 0..1  : which 16-row tile
    const int ncol  = wave & 3;             // 0..3  : which 256-wide o slice
    const int nrow  = lane & 15;            // A row / B col / C col
    const int hi    = lane >> 4;            // K-half selector
    const unsigned short* arow = xt + (mtile * 16 + nrow) * XROW;
    const int khA = hi * 8;                 // A: lo lanes K0..7,16..23 ; hi lanes K8..15,24..31

    float sums[8] = {0.f,0.f,0.f,0.f,0.f,0.f,0.f,0.f};

    for (int t = 0; t < 16; ++t) {
        const int obase = ncol * 256 + t * 16;
        // B[k,n] = W[obase+n, k]; lo lanes K0..7,16..23 ; hi lanes K8..15,24..31 per step
        const unsigned short* brow = Wb + (long)(obase + nrow) * 1024 + hi * 8;
        v8f c = {};

        Frag aA, bA, aB, bB;
        aA.u[0] = *(const uint4*)(arow + khA);
        aA.u[1] = *(const uint4*)(arow + 16 + khA);
        bA.u[0] = *(const uint4*)(brow);
        bA.u[1] = *(const uint4*)(brow + 16);

        #pragma unroll 4
        for (int k0 = 0; k0 < 1024; k0 += 64) {
            // prefetch step k0+32 while WMMA(k0) runs
            aB.u[0] = *(const uint4*)(arow + k0 + 32 + khA);
            aB.u[1] = *(const uint4*)(arow + k0 + 48 + khA);
            bB.u[0] = *(const uint4*)(brow + k0 + 32);
            bB.u[1] = *(const uint4*)(brow + k0 + 48);
            c = __builtin_amdgcn_wmma_f32_16x16x32_bf16(
                    false, aA.v, false, bA.v, (short)0, c, false, false);

            // prefetch step k0+64 (last iter over-reads 32 B; slack/unused)
            aA.u[0] = *(const uint4*)(arow + k0 + 64 + khA);
            aA.u[1] = *(const uint4*)(arow + k0 + 80 + khA);
            bA.u[0] = *(const uint4*)(brow + k0 + 64);
            bA.u[1] = *(const uint4*)(brow + k0 + 80);
            c = __builtin_amdgcn_wmma_f32_16x16x32_bf16(
                    false, aB.v, false, bB.v, (short)0, c, false, false);
        }

        // C layout: lane holds col N=nrow, rows M = i + 8*hi  (i = VGPR idx)
        const float bb = bias[obase + nrow];
        const float kk = keyv[obase + nrow];
        #pragma unroll
        for (int i = 0; i < 8; ++i)
            sums[i] += kk * fast_tanh(c[i] + bb);
    }

    // Reduce the 16 N-lanes of each half-wave; lanes 0 / 16 commit rows.
    #pragma unroll
    for (int i = 0; i < 8; ++i) {
        float s = sums[i];
        s += __shfl_xor(s, 8, 32);
        s += __shfl_xor(s, 4, 32);
        s += __shfl_xor(s, 2, 32);
        s += __shfl_xor(s, 1, 32);
        if (nrow == 0)
            atomicAdd(&scores[rowbase + mtile * 16 + hi * 8 + i], s);
    }
}

// ---------------------------------------------------------- masked softmax
__global__ void __launch_bounds__(256) softmax_kernel(
    const float* __restrict__ scores,
    const long long* __restrict__ lengths,
    float* __restrict__ attn)
{
    __shared__ float red[256];
    const int b = blockIdx.x;
    const int tid = threadIdx.x;
    const long long len = lengths[b];
    const float* srow = scores + b * 2048;

    float v[8];
    float mx = -3.0e38f;
    #pragma unroll
    for (int j = 0; j < 8; ++j) {
        const int s = tid + j * 256;
        const float val = ((long long)s < len) ? srow[s] : -1.0e9f;
        v[j] = val;
        mx = fmaxf(mx, val);
    }
    red[tid] = mx; __syncthreads();
    for (int off = 128; off > 0; off >>= 1) {
        if (tid < off) red[tid] = fmaxf(red[tid], red[tid + off]);
        __syncthreads();
    }
    mx = red[0]; __syncthreads();

    float sum = 0.f;
    #pragma unroll
    for (int j = 0; j < 8; ++j) { v[j] = __expf(v[j] - mx); sum += v[j]; }
    red[tid] = sum; __syncthreads();
    for (int off = 128; off > 0; off >>= 1) {
        if (tid < off) red[tid] += red[tid + off];
        __syncthreads();
    }
    const float inv = 1.0f / red[0];
    #pragma unroll
    for (int j = 0; j < 8; ++j)
        attn[b * 2048 + tid + j * 256] = v[j] * inv;
}

// ---------------------------------------------- out[b,h] = sum_s attn * x
__global__ void __launch_bounds__(256) wsum_kernel(
    const float* __restrict__ x,
    const float* __restrict__ attn,
    float* __restrict__ out)
{
    __shared__ float aw[128];
    const int b   = blockIdx.x;
    const int s0  = blockIdx.y * 128;
    const int tid = threadIdx.x;
    if (tid < 128) aw[tid] = attn[b * 2048 + s0 + tid];
    __syncthreads();

    const int h = tid * 4;
    const float* xp = x + ((long)b * 2048 + s0) * 1024 + h;
    float4 acc = make_float4(0.f, 0.f, 0.f, 0.f);
    #pragma unroll 4
    for (int s = 0; s < 128; ++s) {
        const float w = aw[s];
        const float4 xv = *(const float4*)(xp + (long)s * 1024);
        acc.x += w * xv.x; acc.y += w * xv.y;
        acc.z += w * xv.z; acc.w += w * xv.w;
    }
    float* op = out + b * 1024 + h;
    atomicAdd(op + 0, acc.x);
    atomicAdd(op + 1, acc.y);
    atomicAdd(op + 2, acc.z);
    atomicAdd(op + 3, acc.w);
}

// ------------------------------------------------------------------ launch
extern "C" void kernel_launch(void* const* d_in, const int* in_sizes, int n_in,
                              void* d_out, int out_size, void* d_ws, size_t ws_size,
                              hipStream_t stream) {
    const float*     x       = (const float*)d_in[0];      // [16,2048,1024]
    const long long* lengths = (const long long*)d_in[1];  // [16] int64
    const float*     W       = (const float*)d_in[2];      // [1024,1024]
    const float*     bias    = (const float*)d_in[3];      // [1024]
    const float*     keyv    = (const float*)d_in[4];      // [1024]
    float* out = (float*)d_out;                            // [16,1024]

    unsigned short* Wb = (unsigned short*)d_ws;                       // 2 MB
    float* scores = (float*)((char*)d_ws + (2u << 20));               // 128 KB
    float* attn   = scores + 16 * 2048;                               // 128 KB

    wconv_kernel<<<1024, 256, 0, stream>>>(W, Wb);
    zero_kernel<<<64, 256, 0, stream>>>(out, 16 * 1024);
    zero_kernel<<<128, 256, 0, stream>>>(scores, 16 * 2048);

    fused_scores_kernel<<<1024, 256, 32 * XROW * 2 + 256, stream>>>(x, Wb, bias, keyv, scores);
    softmax_kernel<<<16, 256, 0, stream>>>(scores, lengths, attn);
    wsum_kernel<<<dim3(16, 16), 256, 0, stream>>>(x, attn, out);
}
